// Koop_space_61641370632650
// MI455X (gfx1250) — matrix operand order repaired and checked by reading, not compiled, not run on hardware
//
#include <hip/hip_runtime.h>
#include <cmath>

#define DD 256
#define BB 1024
#define TT 512

typedef __attribute__((ext_vector_type(8)))  float   v8f;
typedef __attribute__((ext_vector_type(16))) __bf16  v16bf;
typedef __attribute__((ext_vector_type(8)))  __bf16  v8bf;
typedef __attribute__((ext_vector_type(4)))  __bf16  v4bf;

static constexpr int LDW = 40;   // weight panel row stride (halves)

// ---------------------------------------------------------------------------
// WMMA helper (CDNA5: D = A(16x32 bf16) * B(32x16 bf16) + C(16x16 f32))
// ---------------------------------------------------------------------------
static __device__ inline v8f wmma_bf16(v16bf a, v16bf b, v8f c) {
    return __builtin_amdgcn_wmma_f32_16x16x32_bf16(
        /*neg_a=*/false, a, /*neg_b=*/false, b,
        /*c_mod=*/(short)0, c, /*reuse_a=*/false, /*reuse_b=*/false);
}

static __device__ inline v16bf cat16(v8bf lo, v8bf hi) {
    return __builtin_shufflevector(lo, hi, 0,1,2,3,4,5,6,7,8,9,10,11,12,13,14,15);
}

// 16 contiguous f32 -> v16bf (for hoisted B-fragments)
static __device__ inline v16bf cvt16_f32(const float* p) {
    v16bf f;
#pragma unroll
    for (int j = 0; j < 16; j += 4) {
        float4 a = ((const float4*)(p + j))[0];
        f[j]   = (__bf16)a.x; f[j+1] = (__bf16)a.y;
        f[j+2] = (__bf16)a.z; f[j+3] = (__bf16)a.w;
    }
    return f;
}

// A-fragment (16x32 bf16) from bf16 LDS; p = base + row*ldx + k0 + (lane>>4)*16.
// Lane holds K = {base..base+15}.
static __device__ inline v16bf afrag_bf16(const __bf16* p) {
    v8bf lo = ((const v8bf*)p)[0];
    v8bf hi = ((const v8bf*)(p + 8))[0];
    return cat16(lo, hi);
}

// B-fragment (32x16 bf16) from bf16 LDS; p = base + col*ldw + (lane>>4)*16.
static __device__ inline v16bf bfrag16(const __bf16* p) {
    v8bf lo = ((const v8bf*)p)[0];
    v8bf hi = ((const v8bf*)p)[1];
    return cat16(lo, hi);
}

// ---------------------------------------------------------------------------
// Double-buffered weight-panel staging (global f32 -> regs -> LDS bf16)
// Panel: NR rows x 32 k, NR in {256,512}; CNT = NR/32 float4 per thread.
// ---------------------------------------------------------------------------
template<int CNT>
static __device__ inline void panel_load(float4* st, const float* __restrict__ W,
                                         int ldg, int k0, int tid) {
#pragma unroll
    for (int q = 0; q < CNT; ++q) {
        int e = tid + q * 256;
        int n = e >> 3, c4 = e & 7;
        st[q] = ((const float4*)(W + (size_t)n * ldg + k0))[c4];
    }
}

template<int CNT>
static __device__ inline void panel_store(const float4* st, __bf16* dst, int tid) {
#pragma unroll
    for (int q = 0; q < CNT; ++q) {
        int e = tid + q * 256;
        int n = e >> 3, c4 = e & 7;
        v4bf d;
        d[0] = (__bf16)st[q].x; d[1] = (__bf16)st[q].y;
        d[2] = (__bf16)st[q].z; d[3] = (__bf16)st[q].w;
        *((v4bf*)(dst + n * LDW + c4 * 4)) = d;
    }
}

// ===========================================================================
// Kernel 1: sequential scan  ks <- ks @ A_w^T + u_t (x) b_vec,  write KS_pre
// 64 blocks x 256 threads; each block owns 16 batch rows for all 512 steps.
//  - A_w^T fragments hoisted into registers (loop-invariant over t).
//  - State kept in LDS as bf16 (WMMA consumes bf16 anyway; f32 accumulator
//    still written to KS_pre), so A-fragment = 2x ds_load_b128, zero cvt.
//  - A-fragment software-pipelined (a_cur/a_nxt) so new fragment loads land
//    in registers the in-flight WMMA doesn't read (kills WAR v_nops).
// ===========================================================================
__global__ __launch_bounds__(256, 1)
void scan_kernel(const float* __restrict__ KS0, const float* __restrict__ VF,
                 const float* __restrict__ Aw,  const float* __restrict__ Bw,
                 float* __restrict__ KSpre)
{
    constexpr int LDXH = DD + 8;  // 264 halves (bf16 state row stride)
    constexpr int LDU  = 24;      // u chunk row stride (floats)

    __shared__ __align__(16) __bf16 ks_bf[16 * LDXH];  // 8.4 KB
    __shared__ __align__(16) float  u_sm[64 * LDU];    // 6.1 KB

    const int tid  = threadIdx.x;
    const int lane = tid & 31;
    const int w    = tid >> 5;
    const int r    = lane & 15;
    const int h    = lane >> 4;
    const int m0   = blockIdx.x * 16;
    const int n0b  = w * 32;          // each wave owns 2 column tiles of 16

    // --- hoist A_w^T fragments (B operand) straight from global into regs ---
    v16bf bfr[8][2];
#pragma unroll
    for (int kb = 0; kb < 8; ++kb)
#pragma unroll
        for (int ct = 0; ct < 2; ++ct)
            bfr[kb][ct] = cvt16_f32(Aw + (size_t)(n0b + ct * 16 + r) * DD
                                       + kb * 32 + h * 16);

    float bv[2];
#pragma unroll
    for (int ct = 0; ct < 2; ++ct) bv[ct] = Bw[n0b + ct * 16 + r];

    // --- ks0 -> ks_bf (bf16) and KS_pre[0] (f32 copy) ---
    for (int q = 0; q < 4; ++q) {
        int e = tid + q * 256;               // float4 index over 16x256
        int row = e >> 6, c4 = e & 63;
        float4 v = ((const float4*)(KS0 + (size_t)(m0 + row) * DD))[c4];
        ((float4*)(KSpre + (size_t)(m0 + row) * DD))[c4] = v;
        v4bf d;
        d[0] = (__bf16)v.x; d[1] = (__bf16)v.y;
        d[2] = (__bf16)v.z; d[3] = (__bf16)v.w;
        *((v4bf*)(ks_bf + row * LDXH + c4 * 4)) = d;
    }

    for (int t = 0; t < TT; ++t) {
        if ((t & 63) == 0) {                 // stage next 64 u columns in LDS
            __syncthreads();
            for (int q = 0; q < 4; ++q) {
                int e = tid + q * 256;       // 16 rows x 64 steps
                int row = e >> 6, j = e & 63;
                u_sm[j * LDU + row] = VF[(size_t)(m0 + row) * TT + t + j];
            }
        }
        __syncthreads();                     // ks_bf / u_sm ready

        v8f z = {0.f,0.f,0.f,0.f,0.f,0.f,0.f,0.f};
        v8f acc[2] = {z, z};

        const __bf16* abase = ks_bf + r * LDXH + h * 16;
        v16bf a_cur = afrag_bf16(abase);
#pragma unroll
        for (int kb = 0; kb < 8; ++kb) {
            v16bf a_nxt = (kb < 7) ? afrag_bf16(abase + (kb + 1) * 32) : a_cur;
            acc[0] = wmma_bf16(a_cur, bfr[kb][0], acc[0]);
            acc[1] = wmma_bf16(a_cur, bfr[kb][1], acc[1]);
            a_cur = a_nxt;
        }

        // rank-1 update: acc[m,n] += u_t[m] * b_vec[n]
        const float* up = u_sm + (t & 63) * LDU + h * 8;
        float4 u0 = ((const float4*)up)[0];
        float4 u1 = ((const float4*)up)[1];
        float uu[8] = {u0.x,u0.y,u0.z,u0.w,u1.x,u1.y,u1.z,u1.w};
#pragma unroll
        for (int ct = 0; ct < 2; ++ct)
#pragma unroll
            for (int i = 0; i < 8; ++i)
                acc[ct][i] += uu[i] * bv[ct];

        __syncthreads();                     // everyone done reading ks_bf

        float* gbase = KSpre + ((size_t)(t + 1) * BB + m0) * DD;
#pragma unroll
        for (int ct = 0; ct < 2; ++ct) {
            int n = n0b + ct * 16 + r;
#pragma unroll
            for (int i = 0; i < 8; ++i) {
                int m = h * 8 + i;           // D-layout: vgpr i -> row
                float v = acc[ct][i];
                ks_bf[m * LDXH + n] = (__bf16)v;      // next-step state (bf16)
                gbase[(size_t)m * DD + n] = v;        // full-precision output
            }
        }
    }
}

// ===========================================================================
// Kernel 2: fused MLP over 513*1024 tokens, 64 tokens per workgroup.
// Weight k-panels: double-buffered LDS with register staging so global-load
// latency overlaps WMMA compute of the previous panel.
// ===========================================================================
__global__ __launch_bounds__(256, 1)
void mlp_kernel(const float* __restrict__ KSpre,
                const float* __restrict__ W0, const float* __restrict__ b0,
                const float* __restrict__ W1, const float* __restrict__ b1,
                const float* __restrict__ lng, const float* __restrict__ lnb,
                const float* __restrict__ W2, const float* __restrict__ b2,
                const float* __restrict__ W3, const float* __restrict__ b3,
                float* __restrict__ OS)
{
    constexpr int LDA  = DD + 4;       // 260 floats  (x / y rows)
    constexpr int LDH  = 2 * DD + 8;   // 520 halves  (s1 rows)
    constexpr int LDS2 = DD + 8;       // 264 halves  (xbf / s2 / s3 rows)
    constexpr int WSZ  = 512 * LDW;    // one panel buffer (halves)

    __shared__ __align__(16) char   bufA_raw[64 * LDA * 4];  // x(f32) -> s2(bf16)
    __shared__ __align__(16) char   bufB_raw[64 * LDH * 2];  // s1 -> y(f32) -> s3
    __shared__ __align__(16) __bf16 xbf[64 * LDS2];          // bf16 copy of X
    __shared__ __align__(16) __bf16 wsm[2 * WSZ];            // double-buffered panel
    __shared__ float W3sm[3 * DD];

    float*  xA   = (float*)bufA_raw;
    __bf16* s2sm = (__bf16*)bufA_raw;
    __bf16* s1sm = (__bf16*)bufB_raw;
    float*  ysm  = (float*)bufB_raw;
    __bf16* s3sm = (__bf16*)bufB_raw;

    const int tid  = threadIdx.x;
    const int lane = tid & 31;
    const int w    = tid >> 5;
    const int r    = lane & 15;
    const int h    = lane >> 4;
    const size_t t0 = (size_t)blockIdx.x * 64;   // flat token row base

    // ---- phase 0: load X tile (64x256), keep f32 + bf16 copies; load W3 ----
    for (int q = 0; q < 16; ++q) {
        int e = tid + q * 256;
        int row = e >> 6, c4 = e & 63;
        float4 v = ((const float4*)(KSpre + (t0 + row) * DD))[c4];
        ((float4*)(xA + row * LDA))[c4] = v;
        v4bf d;
        d[0] = (__bf16)v.x; d[1] = (__bf16)v.y;
        d[2] = (__bf16)v.z; d[3] = (__bf16)v.w;
        *((v4bf*)(xbf + row * LDS2 + c4 * 4)) = d;
    }
    for (int e = tid; e < 3 * DD; e += 256) W3sm[e] = W3[e];

    v8f z = {0.f,0.f,0.f,0.f,0.f,0.f,0.f,0.f};

    // ---- GEMM1: s1 = tanh(X @ W0^T + b0), 64x512, K=256 ----
    const int rt1 = (w & 3) * 16;
    const int cg1 = (w >> 2) * 256;
    v8f acc1[16];
#pragma unroll
    for (int ct = 0; ct < 16; ++ct) acc1[ct] = z;
    {
        float4 st[16];
        panel_load<16>(st, W0, DD, 0, tid);
        for (int kb = 0; kb < 8; ++kb) {
            __syncthreads();                          // prev compute done
            __bf16* buf = wsm + (kb & 1) * WSZ;
            panel_store<16>(st, buf, tid);
            if (kb + 1 < 8) panel_load<16>(st, W0, DD, (kb + 1) * 32, tid);
            __syncthreads();                          // panel visible
            int k0 = kb * 32;
            v16bf a = afrag_bf16(xbf + (rt1 + r) * LDS2 + k0 + h * 16);
#pragma unroll
            for (int ct = 0; ct < 16; ++ct) {
                v16bf b = bfrag16(buf + (cg1 + ct * 16 + r) * LDW + h * 16);
                acc1[ct] = wmma_bf16(a, b, acc1[ct]);
            }
        }
    }
#pragma unroll
    for (int ct = 0; ct < 16; ++ct) {
        int n = cg1 + ct * 16 + r;
        float b0v = b0[n];
#pragma unroll
        for (int i = 0; i < 8; ++i) {
            int m = rt1 + h * 8 + i;
            s1sm[m * LDH + n] = (__bf16)tanhf(acc1[ct][i] + b0v);
        }
    }

    // ---- GEMM2: s2 = tanh(s1 @ W1^T + b1), 64x256, K=512; y = x + s2 ----
    const int rt2 = (w & 3) * 16;
    const int cg2 = (w >> 2) * 128;
    v8f acc2[8];
#pragma unroll
    for (int ct = 0; ct < 8; ++ct) acc2[ct] = z;
    {
        float4 st[8];
        panel_load<8>(st, W1, 2 * DD, 0, tid);
        for (int kb = 0; kb < 16; ++kb) {
            __syncthreads();
            __bf16* buf = wsm + (kb & 1) * WSZ;
            panel_store<8>(st, buf, tid);
            if (kb + 1 < 16) panel_load<8>(st, W1, 2 * DD, (kb + 1) * 32, tid);
            __syncthreads();
            int k0 = kb * 32;
            v16bf a = afrag_bf16(s1sm + (rt2 + r) * LDH + k0 + h * 16);
#pragma unroll
            for (int ct = 0; ct < 8; ++ct) {
                v16bf b = bfrag16(buf + (cg2 + ct * 16 + r) * LDW + h * 16);
                acc2[ct] = wmma_bf16(a, b, acc2[ct]);
            }
        }
    }
    __syncthreads();                          // all s1sm reads done (bufB reuse)
#pragma unroll
    for (int ct = 0; ct < 8; ++ct) {
        int n = cg2 + ct * 16 + r;
        float b1v = b1[n];
#pragma unroll
        for (int i = 0; i < 8; ++i) {
            int m = rt2 + h * 8 + i;
            float s2v = tanhf(acc2[ct][i] + b1v);
            ysm[m * LDA + n] = xA[m * LDA + n] + s2v;
        }
    }
    __syncthreads();

    // ---- LayerNorm rows of y, write bf16 into s2sm (bufA reuse) ----
    {
        int row = tid >> 2, seg = tid & 3;    // 4 lanes per row, in-wave
        const float* yr = ysm + row * LDA + seg * 64;
        float s = 0.f, ss = 0.f;
        for (int j = 0; j < 64; ++j) { float v = yr[j]; s += v; ss += v * v; }
        s  += __shfl_xor(s, 1);  s  += __shfl_xor(s, 2);
        ss += __shfl_xor(ss, 1); ss += __shfl_xor(ss, 2);
        float mean = s * (1.f / 256.f);
        float var  = ss * (1.f / 256.f) - mean * mean;
        float rstd = rsqrtf(var + 1e-5f);
        for (int j = 0; j < 64; ++j) {
            int n = seg * 64 + j;
            float v = (ysm[row * LDA + n] - mean) * rstd;
            s2sm[row * LDS2 + n] = (__bf16)(v * lng[n] + lnb[n]);
        }
    }
    __syncthreads();

    // ---- GEMM3: s3 = elu(s2 @ W2^T + b2), 64x256, K=256 ----
    v8f acc3[8];
#pragma unroll
    for (int ct = 0; ct < 8; ++ct) acc3[ct] = z;
    {
        float4 st[8];
        panel_load<8>(st, W2, DD, 0, tid);
        for (int kb = 0; kb < 8; ++kb) {
            __syncthreads();
            __bf16* buf = wsm + (kb & 1) * WSZ;
            panel_store<8>(st, buf, tid);
            if (kb + 1 < 8) panel_load<8>(st, W2, DD, (kb + 1) * 32, tid);
            __syncthreads();
            int k0 = kb * 32;
            v16bf a = afrag_bf16(s2sm + (rt2 + r) * LDS2 + k0 + h * 16);
#pragma unroll
            for (int ct = 0; ct < 8; ++ct) {
                v16bf b = bfrag16(buf + (cg2 + ct * 16 + r) * LDW + h * 16);
                acc3[ct] = wmma_bf16(a, b, acc3[ct]);
            }
        }
    }
    __syncthreads();
#pragma unroll
    for (int ct = 0; ct < 8; ++ct) {
        int n = cg2 + ct * 16 + r;
        float b2v = b2[n];
#pragma unroll
        for (int i = 0; i < 8; ++i) {
            int m = rt2 + h * 8 + i;
            float vv = acc3[ct][i] + b2v;
            s3sm[m * LDS2 + n] = (__bf16)(vv > 0.f ? vv : (expf(vv) - 1.f));
        }
    }
    __syncthreads();

    // ---- OS = s3 @ W3^T + b3  (3 cols: VALU dot products) ----
    if (tid < 192) {
        int c = tid >> 6, row = tid & 63;
        const __bf16* sr = s3sm + row * LDS2;
        const float*  wr = W3sm + c * DD;
        float accv = 0.f;
        for (int k = 0; k < DD; ++k) accv += (float)sr[k] * wr[k];
        OS[(t0 + row) * 3 + c] = accv + b3[c];
    }
}

// ===========================================================================
extern "C" void kernel_launch(void* const* d_in, const int* in_sizes, int n_in,
                              void* d_out, int out_size, void* d_ws, size_t ws_size,
                              hipStream_t stream)
{
    const float* KS0 = (const float*)d_in[0];
    const float* VF  = (const float*)d_in[1];
    const float* Aw  = (const float*)d_in[2];
    const float* Bw  = (const float*)d_in[3];
    const float* W0  = (const float*)d_in[4];
    const float* b0  = (const float*)d_in[5];
    const float* W1  = (const float*)d_in[6];
    const float* b1  = (const float*)d_in[7];
    const float* lng = (const float*)d_in[8];
    const float* lnb = (const float*)d_in[9];
    const float* W2  = (const float*)d_in[10];
    const float* b2  = (const float*)d_in[11];
    const float* W3  = (const float*)d_in[12];
    const float* b3  = (const float*)d_in[13];

    float* KSpre = (float*)d_out;                        // (513,1024,256)
    float* OS    = KSpre + (size_t)(TT + 1) * BB * DD;   // (513,1024,3)

    scan_kernel<<<BB / 16, 256, 0, stream>>>(KS0, VF, Aw, Bw, KSpre);
    mlp_kernel<<<(TT + 1) * BB / 64, 256, 0, stream>>>(
        KSpre, W0, b0, W1, b1, lng, lnb, W2, b2, W3, b3, OS);
}
